// BilinearMLPEmbedPheno_64364379898013
// MI455X (gfx1250) — matrix operand order repaired and checked by reading, not compiled, not run on hardware
//
#include <hip/hip_runtime.h>
#include <hip/hip_bf16.h>
#include <math.h>

typedef __attribute__((ext_vector_type(16))) _Float16 v16h;
typedef __attribute__((ext_vector_type(8)))  _Float16 h8;
typedef __attribute__((ext_vector_type(8)))  float    v8f;

#define TM      128   // batch rows per block in pheno kernel
#define LD_STR  520   // LDS row stride in halves (512 + 8 pad; keeps 16B align, spreads banks)
#define NTC     4     // N tiles per chunk

__device__ __forceinline__ float gelu_exact(float x) {
    return 0.5f * x * (1.0f + erff(x * 0.70710678118654752f));
}

// ---------------------------------------------------------------------------
// Kernel 1: repack Wp2 (H x H, f32, row-major) into f16 WMMA-B fragments,
// CHUNK-MAJOR: chunk = 4 consecutive N-tiles; within a chunk all K-steps are
// contiguous, so one chunk = ksteps*4 KB of contiguous memory (64 KB @ H=512)
// that the pheno kernel DMA-copies to LDS with async-to-LDS loads.
//   dst[(((chunk*ksteps + kt)*NTC + j)*32 + lane)*16 + e]
//   lane L: N = (chunk*NTC+j)*16 + L%16,  K = kt*32 + (L<16?0:8) + (e<8 ? e : 8+e)
// ---------------------------------------------------------------------------
__global__ void prep_wp2_f16(const float* __restrict__ Wp2,
                             _Float16* __restrict__ dst, int H) {
    int idx = blockIdx.x * blockDim.x + threadIdx.x;
    if (idx >= H * H) return;
    int ksteps = H >> 5;
    int e    = idx & 15;
    int lane = (idx >> 4) & 31;
    int frag = idx >> 9;            // 512 elements per fragment
    int j     = frag & (NTC - 1);
    int kt    = (frag >> 2) % ksteps;
    int chunk = frag / (NTC * ksteps);
    int nt = chunk * NTC + j;
    int K = kt * 32 + ((lane < 16) ? 0 : 8) + ((e < 8) ? e : 8 + e);
    int N = nt * 16 + (lane & 15);
    dst[idx] = (_Float16)Wp2[(size_t)K * H + N];
}

// ---------------------------------------------------------------------------
// Kernel 2: embedding 2-layer MLP + bilinear interaction + small head.
// One thread per batch row; all tiny weights staged in LDS (broadcast reads).
// Writes z into d_out (pheno kernel later does out[row] += p).
// ---------------------------------------------------------------------------
__global__ __launch_bounds__(256)
void embed_bilinear_kernel(const int* __restrict__ x,
                           const float* __restrict__ emb,
                           const float* __restrict__ W1, const float* __restrict__ b1,
                           const float* __restrict__ W2, const float* __restrict__ b2,
                           const float* __restrict__ Wb, const float* __restrict__ ob,
                           const float* __restrict__ Wc1, const float* __restrict__ bc1,
                           const float* __restrict__ Wc2, const float* __restrict__ bc2,
                           float* __restrict__ out, int B) {
    __shared__ float sWb[16 * 16 * 16];
    __shared__ float sW1[256], sW2[256], sWc1[256];
    __shared__ float sb1[16], sb2[16], sbc1[16], sWc2[16], sob[16];
    __shared__ float sbc2;

    int tid = threadIdx.x;
    for (int i = tid; i < 4096; i += 256) sWb[i] = Wb[i];
    if (tid < 256) { sW1[tid] = W1[tid]; sW2[tid] = W2[tid]; sWc1[tid] = Wc1[tid]; }
    if (tid < 16) {
        sb1[tid] = b1[tid]; sb2[tid] = b2[tid]; sbc1[tid] = bc1[tid];
        sWc2[tid] = Wc2[tid]; sob[tid] = ob[tid];
    }
    if (tid == 0) sbc2 = bc2[0];
    __syncthreads();

    int row = blockIdx.x * 256 + tid;
    if (row >= B) return;

    int i0 = x[(size_t)row * 2 + 0];
    int i1 = x[(size_t)row * 2 + 1];

    float e0[16], e1[16];
    {
        const float4* p0 = (const float4*)(emb + (size_t)i0 * 16);
        const float4* p1 = (const float4*)(emb + (size_t)i1 * 16);
#pragma unroll
        for (int c = 0; c < 4; ++c) {
            float4 q = p0[c];
            e0[4 * c + 0] = q.x; e0[4 * c + 1] = q.y; e0[4 * c + 2] = q.z; e0[4 * c + 3] = q.w;
            q = p1[c];
            e1[4 * c + 0] = q.x; e1[4 * c + 1] = q.y; e1[4 * c + 2] = q.z; e1[4 * c + 3] = q.w;
        }
    }

    // h = relu(e@W1 + b1) @ W2 + b2   (for both slots)
    float t[16], h0[16], h1[16];
    for (int j = 0; j < 16; ++j) {
        float s = sb1[j];
#pragma unroll
        for (int k = 0; k < 16; ++k) s += e0[k] * sW1[k * 16 + j];
        t[j] = fmaxf(s, 0.0f);
    }
    for (int j = 0; j < 16; ++j) {
        float s = sb2[j];
#pragma unroll
        for (int k = 0; k < 16; ++k) s += t[k] * sW2[k * 16 + j];
        h0[j] = s;
    }
    for (int j = 0; j < 16; ++j) {
        float s = sb1[j];
#pragma unroll
        for (int k = 0; k < 16; ++k) s += e1[k] * sW1[k * 16 + j];
        t[j] = fmaxf(s, 0.0f);
    }
    for (int j = 0; j < 16; ++j) {
        float s = sb2[j];
#pragma unroll
        for (int k = 0; k < 16; ++k) s += t[k] * sW2[k * 16 + j];
        h1[j] = s;
    }

    // z[q] = ob[q] + sum_p (Wb[q,p,:] . h0) * (Wb[p,q,:] . h1)
    float zq[16];
    for (int q = 0; q < 16; ++q) {
        float accq = sob[q];
        for (int p = 0; p < 16; ++p) {
            const float* w0 = &sWb[(q * 16 + p) * 16];
            const float* w1 = &sWb[(p * 16 + q) * 16];
            float d0 = 0.0f, d1 = 0.0f;
#pragma unroll
            for (int k = 0; k < 16; ++k) { d0 += w0[k] * h0[k]; d1 += w1[k] * h1[k]; }
            accq += d0 * d1;
        }
        zq[q] = accq;
    }

    // head: relu(z@Wc1 + bc1) @ Wc2 + bc2
    float zs = sbc2;
    for (int j = 0; j < 16; ++j) {
        float s = sbc1[j];
#pragma unroll
        for (int q = 0; q < 16; ++q) s += zq[q] * sWc1[q * 16 + j];
        zs += fmaxf(s, 0.0f) * sWc2[j];
    }

    out[row] = zs;   // pheno kernel adds p afterwards
}

// ---------------------------------------------------------------------------
// Kernel 3: pheno MLP with async-LDS double-buffered B staging.
//   Phase 1: g1 = gelu(ph@Wp1) as f16 in LDS.
//   Phase 2: per N-chunk (4 tiles x K=H), DMA the 64KB chunk-contiguous B
//   panel into LDS with GLOBAL_LOAD_ASYNC_TO_LDS_B128 (ASYNCcnt), double
//   buffered so chunk c+1 streams while chunk c runs on the WMMA pipe.
//   Fused epilogue: gelu -> dot Wp3 -> shfl row-reduce -> out[row] += p+bp3.
// ---------------------------------------------------------------------------
__global__ __launch_bounds__(256)
void pheno_mlp_kernel(const float* __restrict__ phenos,
                      const float* __restrict__ Wp1,
                      const _Float16* __restrict__ Wp2f,
                      const float* __restrict__ Wp3,
                      const float* __restrict__ bp3,
                      float* __restrict__ out, int B, int H) {
    extern __shared__ char smem[];
    _Float16* g1   = (_Float16*)smem;                                  // TM * LD_STR halves
    float*    swp3 = (float*)(smem + (size_t)TM * LD_STR * sizeof(_Float16)); // H floats
    float*    sph  = swp3 + H;                                         // TM*2 floats
    _Float16* bbuf = (_Float16*)(sph + TM * 2);                        // 2 x (ksteps*NTC KB)

    const int tid = threadIdx.x;
    const int m0  = blockIdx.x * TM;
    const int ntiles  = H >> 4;                 // 32
    const int ksteps  = H >> 5;                 // 16
    const int nchunks = ntiles / NTC;           // 8
    const int CHB     = ksteps * NTC * 32 * 16 * 2;   // chunk bytes (64KB @ H=512)
    const int CHH     = CHB / 2;                      // chunk halves

    for (int i = tid; i < H; i += 256) swp3[i] = Wp3[i];
    for (int i = tid; i < TM * 2; i += 256) {
        int r = m0 + (i >> 1);
        sph[i] = (r < B) ? phenos[(size_t)r * 2 + (i & 1)] : 0.0f;
    }
    __syncthreads();

    // Phase 1: layer-1 (rank-2 GEMV) + exact gelu -> f16 in LDS
    for (int idx = tid; idx < TM * H; idx += 256) {
        int m = idx / H, n = idx - m * H;
        float pre = sph[2 * m] * Wp1[n] + sph[2 * m + 1] * Wp1[H + n];
        g1[m * LD_STR + n] = (_Float16)gelu_exact(pre);
    }

    // LDS byte offsets of the two B staging buffers (generic-pointer low 32
    // bits == LDS offset per the flat-aperture rule).
    const unsigned int boff0 = (unsigned int)(size_t)(void*)bbuf;
    // Issue the async DMA of chunk 0 into buffer 0 (16 b128 copies / thread).
    {
        const char* gp = (const char*)Wp2f + (size_t)tid * 16;
        unsigned int loff = boff0 + (unsigned int)tid * 16u;
        for (int i = 0; i < CHB / (256 * 16); ++i) {
            asm volatile("global_load_async_to_lds_b128 %0, %1, off"
                         :: "v"(loff), "v"((unsigned long long)(size_t)gp)
                         : "memory");
            gp   += 256 * 16;
            loff += 256 * 16;
        }
    }

    const int wave = tid >> 5;
    const int lane = tid & 31;
    const int ncol = lane & 15;
    const int hiA  = (lane < 16) ? 0 : 8;       // A-frag / C-frag row offset
    const _Float16* arow = g1 + (size_t)(wave * 16 + ncol) * LD_STR;

    float psum[8];
#pragma unroll
    for (int v = 0; v < 8; ++v) psum[v] = 0.0f;

    for (int c = 0; c < nchunks; ++c) {
        // Prefetch next chunk into the other buffer while this one computes.
        if (c + 1 < nchunks) {
            const char* gp = (const char*)Wp2f + (size_t)(c + 1) * CHB + (size_t)tid * 16;
            unsigned int loff = boff0 + (unsigned int)(((c + 1) & 1) * CHB)
                              + (unsigned int)tid * 16u;
            for (int i = 0; i < CHB / (256 * 16); ++i) {
                asm volatile("global_load_async_to_lds_b128 %0, %1, off"
                             :: "v"(loff), "v"((unsigned long long)(size_t)gp)
                             : "memory");
                gp   += 256 * 16;
                loff += 256 * 16;
            }
            // 16 of our own async ops may still be in flight for chunk c+1;
            // async loads retire in order, so <=16 outstanding means chunk c
            // (and phase-1 DS stores, via the barrier below) are complete.
            asm volatile("s_wait_asynccnt 0x10" ::: "memory");
        } else {
            asm volatile("s_wait_asynccnt 0x0" ::: "memory");
        }
        __syncthreads();   // publish buf[c&1] (all waves' slices) + g1 on c==0

        const _Float16* bch = bbuf + (size_t)(c & 1) * CHH;
        v8f acc[NTC];
#pragma unroll
        for (int j = 0; j < NTC; ++j) acc[j] = {};

        for (int kt = 0; kt < ksteps; ++kt) {
            // A fragment (16x32 f16) from LDS: two contiguous 16B runs per lane
            int k0 = kt * 32 + hiA;
            h8 alo = *(const h8*)(arow + k0);
            h8 ahi = *(const h8*)(arow + k0 + 16);
            v16h a = __builtin_shufflevector(alo, ahi,
                     0, 1, 2, 3, 4, 5, 6, 7, 8, 9, 10, 11, 12, 13, 14, 15);

            const _Float16* bbase = bch + ((size_t)(kt * NTC) * 32 + lane) * 16;
#pragma unroll
            for (int j = 0; j < NTC; ++j) {
                v16h b = *(const v16h*)(bbase + (size_t)j * 512);
                acc[j] = __builtin_amdgcn_wmma_f32_16x16x32_f16(
                    false, a, false, b, (short)0, acc[j], false, false);
            }
        }

        // fused epilogue: gelu then fold in Wp3 column weights (per-lane)
        int ntb = c * NTC;
#pragma unroll
        for (int j = 0; j < NTC; ++j) {
            float wn = swp3[(ntb + j) * 16 + ncol];
#pragma unroll
            for (int v = 0; v < 8; ++v)
                psum[v] += gelu_exact(acc[j][v]) * wn;
        }

        __syncthreads();   // all waves done with buf[c&1] before chunk c+2 DMA
    }

    // reduce across the 16 N-lanes of each half-wave; lanes 0/16 commit rows
    float bias = bp3[0];
#pragma unroll
    for (int v = 0; v < 8; ++v) {
        float s = psum[v];
        s += __shfl_xor(s, 1, 32);
        s += __shfl_xor(s, 2, 32);
        s += __shfl_xor(s, 4, 32);
        s += __shfl_xor(s, 8, 32);
        int row = m0 + wave * 16 + hiA + v;
        if (((lane & 15) == 0) && row < B)
            out[row] += s + bias;   // adds onto z written by embed kernel
    }
}

// ---------------------------------------------------------------------------
// Host launcher
// ---------------------------------------------------------------------------
extern "C" void kernel_launch(void* const* d_in, const int* in_sizes, int n_in,
                              void* d_out, int out_size, void* d_ws, size_t ws_size,
                              hipStream_t stream) {
    const int*   x      = (const int*)d_in[0];
    const float* phenos = (const float*)d_in[1];
    const float* emb    = (const float*)d_in[2];
    const float* W1     = (const float*)d_in[3];
    const float* b1     = (const float*)d_in[4];
    const float* W2     = (const float*)d_in[5];
    const float* b2     = (const float*)d_in[6];
    const float* Wb     = (const float*)d_in[7];
    const float* ob     = (const float*)d_in[8];
    const float* Wc1    = (const float*)d_in[9];
    const float* bc1    = (const float*)d_in[10];
    const float* Wc2    = (const float*)d_in[11];
    const float* bc2    = (const float*)d_in[12];
    const float* Wp1    = (const float*)d_in[13];
    const float* Wp2    = (const float*)d_in[14];
    const float* Wp3    = (const float*)d_in[15];
    const float* bp3    = (const float*)d_in[16];
    float* out = (float*)d_out;

    const int B = in_sizes[0] / 2;   // x is (B, 2)
    const int H = in_sizes[15];      // Wp3 is (H, 1)

    _Float16* wp2f = (_Float16*)d_ws;  // H*H*2 bytes = 512 KB

    int nelem = H * H;
    prep_wp2_f16<<<(nelem + 255) / 256, 256, 0, stream>>>(Wp2, wp2f, H);

    embed_bilinear_kernel<<<(B + 255) / 256, 256, 0, stream>>>(
        x, emb, W1, b1, W2, b2, Wb, ob, Wc1, bc1, Wc2, bc2, out, B);

    const int ksteps = H >> 5;
    const size_t chb = (size_t)ksteps * NTC * 32 * 16 * 2;   // 64KB @ H=512
    size_t smem = (size_t)TM * LD_STR * sizeof(_Float16)
                + (size_t)H * sizeof(float)
                + (size_t)TM * 2 * sizeof(float)
                + 2 * chb;
    pheno_mlp_kernel<<<(B + TM - 1) / TM, 256, smem, stream>>>(
        phenos, Wp1, wp2f, Wp3, bp3, out, B, H);
}
